// TTTBase_16655883174379
// MI455X (gfx1250) — matrix operand order repaired and checked by reading, not compiled, not run on hardware
//
#include <hip/hip_runtime.h>
#include <hip/hip_bf16.h>

// ---------------------------------------------------------------------------
// TTT preprocessing for MI455X (gfx1250, wave32, WMMA).
// B=4, L=2048, C=2048, H=32, D=64, MB=16, NM=128.
// Outputs (flat, in order): XQ | XK | XV (each B*H*NM*MB*D = 16777216 f32),
// then eta (B*H*NM*MB*MB = 4194304 f32).
// GEMMs run on v_wmma_f32_16x16x32_bf16 with a hi/lo bf16 split (bf16x3)
// for near-fp32 accuracy at matrix-pipe speed.
// ---------------------------------------------------------------------------

typedef __attribute__((ext_vector_type(16))) __bf16 v16bf;
typedef __attribute__((ext_vector_type(8)))  float  v8f;

struct U32x2q { uint4 a, b; };   // 32 bytes == v16bf

static __device__ inline v16bf frag_ld(const unsigned* p, int second_off_uints) {
    U32x2q u;
    u.a = *(const uint4*)(p);
    u.b = *(const uint4*)(p + second_off_uints);
    return __builtin_bit_cast(v16bf, u);
}

// Split two consecutive fp32 values into packed bf16 hi-pair and lo-pair.
// hi = truncate-to-bf16(x) (so lo = x - hi is exact), lo = nearest-bf16(x - hi), ties away from zero.
static __device__ inline void split2(float x0, float x1, unsigned& hi, unsigned& lo) {
    unsigned u0 = __float_as_uint(x0), u1 = __float_as_uint(x1);
    unsigned h0 = u0 & 0xFFFF0000u,    h1 = u1 & 0xFFFF0000u;
    float r0 = x0 - __uint_as_float(h0);
    float r1 = x1 - __uint_as_float(h1);
    unsigned l0 = (__float_as_uint(r0) + 0x8000u) >> 16;
    unsigned l1 = (__float_as_uint(r1) + 0x8000u) & 0xFFFF0000u;
    hi = (h0 >> 16) | h1;    // k even -> [15:0], k odd -> [31:16]
    lo = l0 | l1;
}

#define BMT   64        // tokens per block
#define KC    32        // K chunk (== WMMA bf16 K depth)
#define LDAP  20        // padded LDS row stride in uints (16 data pairs + 4 pad)

__global__ __launch_bounds__(256)
void ttt_qkv_rope_kernel(const float* __restrict__ X,    // (8192, 2048) row-major
                         const float* __restrict__ Wq,   // (2048, 2048) row-major (o, c)
                         const float* __restrict__ Wk,
                         const float* __restrict__ Wv,
                         float* __restrict__ out)        // XQ|XK|XV
{
    __shared__ __align__(16) unsigned sAh[BMT * LDAP];
    __shared__ __align__(16) unsigned sAl[BMT * LDAP];
    __shared__ __align__(16) unsigned sBh[3][BMT * LDAP];
    __shared__ __align__(16) unsigned sBl[3][BMT * LDAP];

    const int tid  = threadIdx.x;
    const int bm   = blockIdx.x;      // token block: 0..127 (64 tokens each)
    const int h    = blockIdx.y;      // head: 0..31 (64 channels each)
    const int lane = tid & 31;
    const int wid  = tid >> 5;        // 0..7 (wave32)
    const int wm   = wid & 1;         // wave row: 2x16 token tiles
    const int wn   = wid >> 1;        // wave col: 16-channel tile
    const int l15  = lane & 15;
    const int half = lane >> 4;

    const float* Wmat[3] = { Wq, Wk, Wv };

    v8f acc[3][2];
#pragma unroll
    for (int w = 0; w < 3; ++w)
#pragma unroll
        for (int t = 0; t < 2; ++t)
            acc[w][t] = v8f{0.f,0.f,0.f,0.f,0.f,0.f,0.f,0.f};

    const int row0 = bm * BMT;        // global token-row base (B*L flattened)

    for (int kc = 0; kc < 2048; kc += KC) {
        // ---- stage A (64x32 f32 -> bf16 hi/lo pairs) ----
#pragma unroll
        for (int p = 0; p < 2; ++p) {
            int idx = tid + p * 256;              // 0..511
            int r = idx >> 3, c4 = idx & 7;       // row 0..63, float4 col 0..7
            float4 v = *(const float4*)(X + (size_t)(row0 + r) * 2048 + kc + c4 * 4);
            unsigned h0, l0, h1, l1;
            split2(v.x, v.y, h0, l0);
            split2(v.z, v.w, h1, l1);
            *(uint2*)&sAh[r * LDAP + c4 * 2] = make_uint2(h0, h1);
            *(uint2*)&sAl[r * LDAP + c4 * 2] = make_uint2(l0, l1);
        }
        // ---- stage B for q/k/v: rows = output channel n (W row h*64+n is contiguous in c) ----
#pragma unroll
        for (int w = 0; w < 3; ++w) {
#pragma unroll
            for (int p = 0; p < 2; ++p) {
                int idx = tid + p * 256;
                int r = idx >> 3, c4 = idx & 7;
                float4 v = *(const float4*)(Wmat[w] + (size_t)(h * 64 + r) * 2048 + kc + c4 * 4);
                unsigned h0, l0, h1, l1;
                split2(v.x, v.y, h0, l0);
                split2(v.z, v.w, h1, l1);
                *(uint2*)&sBh[w][r * LDAP + c4 * 2] = make_uint2(h0, h1);
                *(uint2*)&sBl[w][r * LDAP + c4 * 2] = make_uint2(l0, l1);
            }
        }
        __syncthreads();

        // ---- WMMA: 2 row tiles x 3 outputs x 3 split terms ----
#pragma unroll
        for (int t = 0; t < 2; ++t) {
            // A 16x32 bf16 frag: lane<16 -> K {0..7,16..23}, lane>=16 -> +8
            int m = wm * 32 + t * 16 + l15;
            v16bf ah = frag_ld(&sAh[m * LDAP + 4 * half], 8);
            v16bf al = frag_ld(&sAl[m * LDAP + 4 * half], 8);
#pragma unroll
            for (int w = 0; w < 3; ++w) {
                // B 32x16 bf16 frag: lane<16 -> K {0..7,16..23}, lane>=16 -> +8
                int n = wn * 16 + l15;
                v16bf bh = frag_ld(&sBh[w][n * LDAP + 4 * half], 8);
                v16bf bl = frag_ld(&sBl[w][n * LDAP + 4 * half], 8);
                acc[w][t] = __builtin_amdgcn_wmma_f32_16x16x32_bf16(
                    false, ah, false, bh, (short)0, acc[w][t], false, false);
                acc[w][t] = __builtin_amdgcn_wmma_f32_16x16x32_bf16(
                    false, al, false, bh, (short)0, acc[w][t], false, false);
                acc[w][t] = __builtin_amdgcn_wmma_f32_16x16x32_bf16(
                    false, ah, false, bl, (short)0, acc[w][t], false, false);
            }
        }
        __syncthreads();
    }

    // ---- epilogue: fused RoPE (Q,K) + scatter to (B,H,NM,MB,D) ----
    // Net effect of permute/rotate/undo: pairwise rotation on channels (2p,2p+1)
    // with angle (l % 16) * 10000^(-p/32).  pos == in-tile row (bases are %16==0).
    const int   d   = wn * 16 + l15;            // channel within head
    const int   pp  = d >> 1;
    const float inv = __expf(-(float)pp * (9.210340371976184f / 32.0f)); // 10000^(-p/32)
    const float sgn = (d & 1) ? 1.0f : -1.0f;

#pragma unroll
    for (int t = 0; t < 2; ++t) {
#pragma unroll
        for (int w = 0; w < 3; ++w) {
#pragma unroll
            for (int r = 0; r < 8; ++r) {
                float v = acc[w][t][r];
                int rit = r + 8 * half;          // row in 16-tile == pos == mb
                float o;
                if (w < 2) {
                    float pr = __shfl_xor(v, 1, 32);   // partner channel d^1, same row
                    float s, c;
                    __sincosf((float)rit * inv, &s, &c);
                    o = v * c + sgn * pr * s;
                } else {
                    o = v;
                }
                int gr = row0 + wm * 32 + t * 16 + rit;  // 0..8191
                int b  = gr >> 11;
                int l  = gr & 2047;
                int nm = l >> 4;
                size_t off = (size_t)w * 16777216u +
                             (((((size_t)b * 32 + h) * 128 + nm) * 16 + rit) * 64 + d);
                out[off] = o;
            }
        }
    }
}

// ---------------------------------------------------------------------------
// eta kernel: s[b,h,nm,j] = sigmoid(X4[b,nm,j,:] . lrw[h,:] + lrb[h]) / 64
//             eta[b,h,nm,i,j] = max(1/(i+1)+lti[i], 0) * s[b,h,nm,j]
// One block per (b, nm): 16 tokens x 32 heads, LDS-chunked dot products.
// ---------------------------------------------------------------------------
#define ECK 256

__global__ __launch_bounds__(256)
void ttt_eta_kernel(const float* __restrict__ X,    // (8192, 2048)
                    const float* __restrict__ lrw,  // (32, 2048)
                    const float* __restrict__ lrb,  // (32)
                    const float* __restrict__ lti,  // (16)
                    float* __restrict__ eta)        // (4,32,128,16,16)
{
    __shared__ __align__(16) float Xs[16][ECK];
    __shared__ float Ws[32][ECK + 1];   // +1 pad: conflict-free column reads
    __shared__ float Ss[32][16];
    __shared__ float Ti[16];

    const int tid = threadIdx.x;
    const int blk = blockIdx.x;        // 0..511
    const int b   = blk >> 7;
    const int nm  = blk & 127;
    const int hh  = tid & 31;          // head
    const int jj  = tid >> 5;          // 0..7 (token; also handles jj+8)

    float a0 = 0.f, a1 = 0.f;
    const size_t xbase = ((size_t)b * 2048 + (size_t)nm * 16) * 2048;

    for (int cc = 0; cc < 2048; cc += ECK) {
#pragma unroll
        for (int p = 0; p < 4; ++p) {                 // X chunk: 16 x 256
            int idx = tid + p * 256;                  // 0..1023
            int r = idx >> 6, c = (idx & 63) * 4;
            float4 v = *(const float4*)(X + xbase + (size_t)r * 2048 + cc + c);
            *(float4*)&Xs[r][c] = v;
        }
#pragma unroll
        for (int p = 0; p < 8; ++p) {                 // W chunk: 32 x 256
            int idx = tid + p * 256;                  // 0..2047
            int r = idx >> 6, c = (idx & 63) * 4;
            float4 v = *(const float4*)(lrw + (size_t)r * 2048 + cc + c);
            Ws[r][c] = v.x; Ws[r][c + 1] = v.y; Ws[r][c + 2] = v.z; Ws[r][c + 3] = v.w;
        }
        __syncthreads();
#pragma unroll 4
        for (int k = 0; k < ECK; ++k) {
            float w = Ws[hh][k];
            a0 = fmaf(Xs[jj][k],     w, a0);
            a1 = fmaf(Xs[jj + 8][k], w, a1);
        }
        __syncthreads();
    }

    const float bias = lrb[hh];
    Ss[hh][jj]     = (1.0f / 64.0f) / (1.0f + __expf(-(a0 + bias)));
    Ss[hh][jj + 8] = (1.0f / 64.0f) / (1.0f + __expf(-(a1 + bias)));
    if (tid < 16) Ti[tid] = fmaxf(1.0f / (float)(tid + 1) + lti[tid], 0.0f);
    __syncthreads();

#pragma unroll
    for (int it = 0; it < 32; ++it) {                  // 8192 eta values / block
        int idx = tid + it * 256;
        int hw = idx >> 8;
        int i  = (idx >> 4) & 15;
        int j  = idx & 15;
        eta[((((size_t)b * 32 + hw) * 128 + nm) * 256) + i * 16 + j] = Ti[i] * Ss[hw][j];
    }
}

extern "C" void kernel_launch(void* const* d_in, const int* in_sizes, int n_in,
                              void* d_out, int out_size, void* d_ws, size_t ws_size,
                              hipStream_t stream) {
    const float* X   = (const float*)d_in[0];
    const float* Wq  = (const float*)d_in[1];
    const float* Wk  = (const float*)d_in[2];
    const float* Wv  = (const float*)d_in[3];
    const float* lrw = (const float*)d_in[4];
    const float* lrb = (const float*)d_in[5];
    const float* lti = (const float*)d_in[6];
    float* out = (float*)d_out;

    ttt_qkv_rope_kernel<<<dim3(128, 32), 256, 0, stream>>>(X, Wq, Wk, Wv, out);
    ttt_eta_kernel<<<dim3(512), 256, 0, stream>>>(X, lrw, lrb, lti, out + 50331648);
}
